// SimplifiedEncoder_84628035600922
// MI455X (gfx1250) — matrix operand order repaired and checked by reading, not compiled, not run on hardware
//
#include <hip/hip_runtime.h>
#include <math.h>

// Problem constants (from the reference)
#define VOCAB  32000
#define E_DIM  512
#define H_DIM  1024
#define G_DIM  4096      // 4*H
#define T_LEN  512
#define B_SZ   256

#define BCOL_STRIDE 40   // per-column LDS stride: 32 K-elems + 8 pad elems (16B)

typedef __attribute__((ext_vector_type(16))) __bf16 bf16x16;
typedef __attribute__((ext_vector_type(8)))  __bf16 bf16x8;
typedef __attribute__((ext_vector_type(8)))  float  floatx8;
typedef __attribute__((ext_vector_type(4)))  unsigned int uint32x4;
typedef __attribute__((ext_vector_type(8)))  int int32x8;
typedef __attribute__((ext_vector_type(4)))  int int32x4;

#if defined(__has_builtin)
#  if __has_builtin(__builtin_amdgcn_tensor_load_to_lds)
#    define HAVE_TDM 1
#  else
#    define HAVE_TDM 0
#  endif
#else
#  define HAVE_TDM 0
#endif

// ---------------------------------------------------------------------------
// A-tile loader: 16x32 bf16 A-matrix (row-major source).
// Lanes 0-15: row M=lane, K={0..7,16..23}; lanes 16-31: row M=lane-16,
// K={8..15,24..31}.  Two 16B loads per lane.
// ---------------------------------------------------------------------------
union ATile { bf16x16 v; struct { bf16x8 lo; bf16x8 hi; } p; };

__device__ __forceinline__ bf16x16 load_a_tile(const __bf16* __restrict__ row_base,
                                               int k0, int khalf) {
  ATile t;
  t.p.lo = *(const bf16x8*)(row_base + k0 +      khalf * 8);
  t.p.hi = *(const bf16x8*)(row_base + k0 + 16 + khalf * 8);
  return t.v;
}

#if HAVE_TDM
// ---------------------------------------------------------------------------
// TDM: async-load one B strip (64 weight rows x 32 K-elems, bf16) into LDS.
// D# per CDNA5 ISA ch.8: group0 = {flags, lds_addr, global_addr, type=2},
// group1 = {data_size=2B, pad(16DW interval -> +4DW), dims, strides}.
// Pad turns the 64B column payload into an 80B LDS column stride.
// 6-arg builtin form: (g0, g1, g2, g3, g_extra, cpol); groups 2/3 unused for 2D.
// ---------------------------------------------------------------------------
__device__ __forceinline__ void tdm_load_b_strip(const __bf16* gtile,
                                                 unsigned lds_byte_addr,
                                                 int k_dim) {
  const unsigned long long ga = (unsigned long long)(uintptr_t)gtile;
  uint32x4 g0;
  g0.x = 1u;                                           // count=1 (valid D#)
  g0.y = lds_byte_addr;                                // bits 63:32 lds_addr
  g0.z = (unsigned)(ga & 0xFFFFFFFFu);                 // global_addr[31:0]
  g0.w = (unsigned)((ga >> 32) & 0x01FFFFFFu) | (2u << 30);  // [56:32] | type=2

  const unsigned td0 = (unsigned)k_dim;                // tensor_dim0 (elems)
  const unsigned td1 = (unsigned)G_DIM;                // tensor_dim1 (rows)
  const unsigned long long s0 = (unsigned long long)k_dim;  // dim0 stride (elems)
  int32x8 g1;
  g1[0] = (int)((1u << 16)        // data_size = 2 bytes
              | (1u << 20)        // pad_enable
              | (3u << 22)        // pad_interval: 16 DWORDs (64B) before pad
              | (3u << 25));      // pad_amount:   4 DWORDs (16B)
  g1[1] = (int)((td0 & 0xFFFFu) << 16);                       // tensor_dim0 lo
  g1[2] = (int)((td0 >> 16) | ((td1 & 0xFFFFu) << 16));       // td0 hi | td1 lo
  g1[3] = (int)((td1 >> 16) | (32u << 16));                   // td1 hi | tile_dim0=32
  g1[4] = (int)(64u);                                         // tile_dim1=64, tile_dim2=0
  g1[5] = (int)(s0 & 0xFFFFFFFFull);                          // stride0 lo
  g1[6] = (int)((s0 >> 32) & 0xFFFFull);                      // stride0 hi | stride1 lo=0
  g1[7] = 0;                                                  // stride1 hi
  const int32x4 z4 = {0, 0, 0, 0};
  const int32x8 z8 = {0, 0, 0, 0, 0, 0, 0, 0};
  __builtin_amdgcn_tensor_load_to_lds(g0, g1, z4, z4, z8, 0);
}
#endif

// ---------------------------------------------------------------------------
// One GEMM phase: acc[0..3] (16x64 tile per wave) += A(rows of arow) * Wp^T.
// B strips are double-buffered in LDS, shared by all 4 waves of the block.
// ---------------------------------------------------------------------------
union BTile { bf16x16 v; struct { bf16x8 lo; bf16x8 hi; } p; };

__device__ __forceinline__ void gemm_phase(const __bf16* __restrict__ Wp, int Kp,
                                           const __bf16* __restrict__ arow,
                                           int n0, int wave, int col, int khalf,
                                           __bf16* __restrict__ sm0,
                                           __bf16* __restrict__ sm1,
                                           floatx8* acc)
{
  const int nk = Kp / 32;
  __bf16* bufs[2] = { sm0, sm1 };

#if HAVE_TDM
  if (wave == 0)
    tdm_load_b_strip(Wp + (size_t)n0 * Kp, (unsigned)(uintptr_t)(void*)sm0, Kp);
#endif

  for (int i = 0; i < nk; ++i) {
    const int k0 = i * 32;
    __bf16* cur = bufs[i & 1];

#if HAVE_TDM
    if (wave == 0) {
      if (i + 1 < nk) {
        tdm_load_b_strip(Wp + (size_t)n0 * Kp + (k0 + 32),
                         (unsigned)(uintptr_t)(void*)bufs[(i + 1) & 1], Kp);
        __builtin_amdgcn_s_wait_tensorcnt(1);   // oldest (chunk i) has landed
      } else {
        __builtin_amdgcn_s_wait_tensorcnt(0);   // last chunk: drain
      }
    }
#else
    // Cooperative fill fallback: 128 threads x 2 chunks of 16B = 4KB strip.
    {
      #pragma unroll
      for (int q = 0; q < 2; ++q) {
        const int cid  = (int)threadIdx.x * 2 + q;   // 0..255
        const int cc   = cid >> 2;                   // column 0..63
        const int part = cid & 3;                    // 16B piece within 64B
        *(bf16x8*)(cur + cc * BCOL_STRIDE + part * 8) =
            *(const bf16x8*)(Wp + (size_t)(n0 + cc) * Kp + k0 + part * 8);
      }
    }
#endif
    __syncthreads();                 // strip i visible to all waves

    const bf16x16 a = load_a_tile(arow, k0, khalf);
    #pragma unroll
    for (int j = 0; j < 4; ++j) {
      BTile b;
      const __bf16* p = cur + (j * 16 + col) * BCOL_STRIDE + khalf * 8;
      b.p.lo = *(const bf16x8*)p;
      b.p.hi = *(const bf16x8*)(p + 16);
      acc[j] = __builtin_amdgcn_wmma_f32_16x16x32_bf16(
                 false, a, false, b.v, (short)0, acc[j], false, false);
    }
    __syncthreads();                 // done reading buf before it is refilled
  }
}

// ---------------------------------------------------------------------------
// Per-step gate GEMM: gates[b][g] = bias[g] + emb[x[b]]*Wih[g] + h[b]*Whh[g].
// M=256, N=4096, K=512+1024.  Block = 128 thr (4 waves) -> 64x64 output tile:
// waves split M, share the 64-col B strip via LDS (TDM double buffer).
// grid = (64 N-strips, 4 M-groups).
// ---------------------------------------------------------------------------
__global__ __launch_bounds__(128)
void lstm_gates_wmma(const int*    __restrict__ x_t,
                     const __bf16* __restrict__ emb_bf,   // [VOCAB][E]
                     const __bf16* __restrict__ wih_bf,   // [4H][E]
                     const __bf16* __restrict__ whh_bf,   // [4H][H]
                     const __bf16* __restrict__ h_bf,     // [B][H]
                     const float*  __restrict__ bias,     // [4H]
                     float*        __restrict__ gates)    // [B][4H]
{
  __shared__ __align__(16) __bf16 smem[2][64 * BCOL_STRIDE];   // 10 KB

  const int lane  = threadIdx.x & 31;
  const int wave  = threadIdx.x >> 5;
  const int col   = lane & 15;
  const int khalf = lane >> 4;
  const int n0 = blockIdx.x * 64;
  const int m0 = blockIdx.y * 64 + wave * 16;

  const int row   = m0 + col;
  const int token = x_t[row];
  const __bf16* arow_e = emb_bf + (size_t)token * E_DIM;
  const __bf16* arow_h = h_bf   + (size_t)row   * H_DIM;

  floatx8 zero = {};
  floatx8 acc[4];
  acc[0] = zero; acc[1] = zero; acc[2] = zero; acc[3] = zero;

  gemm_phase(wih_bf, E_DIM, arow_e, n0, wave, col, khalf,
             &smem[0][0], &smem[1][0], acc);
  gemm_phase(whh_bf, H_DIM, arow_h, n0, wave, col, khalf,
             &smem[0][0], &smem[1][0], acc);

  // C/D layout: VGPR r, lane L: M = r + 8*(L>=16), N = L%16
  #pragma unroll
  for (int j = 0; j < 4; ++j) {
    const int n = n0 + j * 16 + col;
    const float bs = bias[n];
    #pragma unroll
    for (int r = 0; r < 8; ++r) {
      const int m = m0 + r + khalf * 8;
      gates[(size_t)m * G_DIM + n] = acc[j][r] + bs;
    }
  }
}

// ---------------------------------------------------------------------------
// Pointwise LSTM cell + pad mask.
// ---------------------------------------------------------------------------
__device__ __forceinline__ float sigmoid_f(float x) {
  return 1.0f / (1.0f + __expf(-x));
}

__global__ __launch_bounds__(256)
void lstm_pointwise(const float* __restrict__ gates,
                    const int*   __restrict__ x_t,
                    float*       __restrict__ h,
                    float*       __restrict__ c,
                    __bf16*      __restrict__ h_bf)
{
  const int idx = blockIdx.x * blockDim.x + threadIdx.x;   // 0 .. B*H-1
  const int b = idx >> 10;
  const int j = idx & (H_DIM - 1);
  if (x_t[b] == 0) return;           // pad: keep previous h, c

  const float* g = gates + (size_t)b * G_DIM;
  const float ig = sigmoid_f(g[j]);
  const float fg = sigmoid_f(g[H_DIM + j]);
  const float gg = tanhf(g[2 * H_DIM + j]);
  const float og = sigmoid_f(g[3 * H_DIM + j]);

  const float cn = fg * c[idx] + ig * gg;
  const float hn = og * tanhf(cn);
  c[idx]    = cn;
  h[idx]    = hn;
  h_bf[idx] = (__bf16)hn;
}

// ---------------------------------------------------------------------------
// Setup / teardown kernels
// ---------------------------------------------------------------------------
__global__ void convert_bf16_kernel(const float* __restrict__ in,
                                    __bf16* __restrict__ out, int n) {
  int i = blockIdx.x * blockDim.x + threadIdx.x;
  const int stride = gridDim.x * blockDim.x;
  for (; i < n; i += stride) out[i] = (__bf16)in[i];
}

__global__ void bias_init_kernel(const float* __restrict__ b_ih,
                                 const float* __restrict__ b_hh,
                                 float* __restrict__ bias) {
  int i = blockIdx.x * blockDim.x + threadIdx.x;
  if (i < G_DIM) bias[i] = b_ih[i] + b_hh[i];
}

__global__ void state_init_kernel(const float* __restrict__ h0,
                                  const float* __restrict__ c0,
                                  float* __restrict__ h, float* __restrict__ c,
                                  __bf16* __restrict__ h_bf) {
  int i = blockIdx.x * blockDim.x + threadIdx.x;
  if (i < B_SZ * H_DIM) {
    const int j = i & (H_DIM - 1);
    const float hv = h0[j];
    h[i] = hv; c[i] = c0[j]; h_bf[i] = (__bf16)hv;
  }
}

__global__ void copy_out_kernel(const float* __restrict__ h,
                                const float* __restrict__ c,
                                float* __restrict__ out) {
  int i = blockIdx.x * blockDim.x + threadIdx.x;
  if (i < B_SZ * H_DIM) {
    out[i] = h[i];
    out[B_SZ * H_DIM + i] = c[i];
  }
}

// ---------------------------------------------------------------------------
// Launch.  Workspace (~50 MB): emb_bf | wih_bf | whh_bf | bias | h | c |
// h_bf | gates.
// ---------------------------------------------------------------------------
extern "C" void kernel_launch(void* const* d_in, const int* in_sizes, int n_in,
                              void* d_out, int out_size, void* d_ws, size_t ws_size,
                              hipStream_t stream) {
  const int*   x    = (const int*)  d_in[0];   // [T][B]
  const float* emb  = (const float*)d_in[1];
  const float* W_ih = (const float*)d_in[2];
  const float* W_hh = (const float*)d_in[3];
  const float* b_ih = (const float*)d_in[4];
  const float* b_hh = (const float*)d_in[5];
  const float* h0   = (const float*)d_in[6];
  const float* c0   = (const float*)d_in[7];

  char* ws = (char*)d_ws;
  size_t off = 0;
  __bf16* emb_bf = (__bf16*)(ws + off); off += (size_t)VOCAB * E_DIM * 2;
  __bf16* wih_bf = (__bf16*)(ws + off); off += (size_t)G_DIM * E_DIM * 2;
  __bf16* whh_bf = (__bf16*)(ws + off); off += (size_t)G_DIM * H_DIM * 2;
  float*  bias   = (float*) (ws + off); off += (size_t)G_DIM * 4;
  float*  h      = (float*) (ws + off); off += (size_t)B_SZ * H_DIM * 4;
  float*  c      = (float*) (ws + off); off += (size_t)B_SZ * H_DIM * 4;
  __bf16* h_bf   = (__bf16*)(ws + off); off += (size_t)B_SZ * H_DIM * 2;
  float*  gates  = (float*) (ws + off);

  convert_bf16_kernel<<<4096, 256, 0, stream>>>(emb,  emb_bf, VOCAB * E_DIM);
  convert_bf16_kernel<<<2048, 256, 0, stream>>>(W_ih, wih_bf, G_DIM * E_DIM);
  convert_bf16_kernel<<<4096, 256, 0, stream>>>(W_hh, whh_bf, G_DIM * H_DIM);
  bias_init_kernel<<<16, 256, 0, stream>>>(b_ih, b_hh, bias);
  state_init_kernel<<<(B_SZ * H_DIM) / 256, 256, 0, stream>>>(h0, c0, h, c, h_bf);

  for (int t = 0; t < T_LEN; ++t) {
    const int* x_t = x + (size_t)t * B_SZ;
    lstm_gates_wmma<<<dim3(64, 4), 128, 0, stream>>>(
        x_t, emb_bf, wih_bf, whh_bf, h_bf, bias, gates);
    lstm_pointwise<<<(B_SZ * H_DIM) / 256, 256, 0, stream>>>(
        gates, x_t, h, c, h_bf);
  }

  copy_out_kernel<<<(B_SZ * H_DIM) / 256, 256, 0, stream>>>(h, c, (float*)d_out);
}
